// GCN_30915174597106
// MI455X (gfx1250) — matrix operand order repaired and checked by reading, not compiled, not run on hardware
//
#include <hip/hip_runtime.h>
#include <hip/hip_bf16.h>
#include <math.h>

// ---------- WMMA types (CDNA5 / gfx1250, wave32) ----------
typedef __bf16 bf16_t;
typedef bf16_t v16bf __attribute__((ext_vector_type(16)));
typedef float  v8f   __attribute__((ext_vector_type(8)));

#define NFEAT 128   // N_FEAT == HIDDEN == 128
#define NGRAPH 64

// ---------------- generic fill ----------------
__global__ void fill_f32(float* __restrict__ p, float v, long n) {
    long i = (long)blockIdx.x * blockDim.x + threadIdx.x;
    long stride = (long)gridDim.x * blockDim.x;
    for (; i < n; i += stride) p[i] = v;
}

// ---------------- degree / dinv ----------------
__global__ void deg_edges(const int* __restrict__ ei, const float* __restrict__ ew,
                          float* __restrict__ deg, int E) {
    int e = blockIdx.x * blockDim.x + threadIdx.x;
    if (e < E) atomicAdd(&deg[ei[E + e]], ew[e]);   // dst = ei[1][e]
}

__global__ void rsqrt_deg(const float* __restrict__ deg, float* __restrict__ dinv, int n) {
    int i = blockIdx.x * blockDim.x + threadIdx.x;
    if (i < n) { float d = deg[i]; dinv[i] = d > 0.f ? rsqrtf(d) : 0.f; }
}

// ---------------- pack weight [128x128] f32 (row-major K x N) into WMMA-B bf16 layout ----
// B (32x16 bf16 per wmma): element i of lane L holds K = (i>>3)*16 + half*8 + (i&7), N = L&15
// packed index: ((kb*8 + colTile)*32 + lane)*16 + elem   (contiguous v16bf per lane)
__global__ void pack_b_bf16(const float* __restrict__ W, const float* __restrict__ W2,
                            bf16_t* __restrict__ out) {
    int g = blockIdx.x * blockDim.x + threadIdx.x;
    if (g >= NFEAT * NFEAT) return;
    int elem =  g        & 15;
    int lane = (g >> 4)  & 31;
    int ct   = (g >> 9)  & 7;
    int kb   =  g >> 12;
    int k = kb * 32 + ((elem >> 3) << 4) + ((lane >> 4) << 3) + (elem & 7);  // (elem>>3)*16 + half*8 + (elem&7)
    int n = ct * 16 + (lane & 15);
    float v = W[k * NFEAT + n];
    if (W2) v += W2[k * NFEAT + n];               // Wlin_top + Wlin_bot fold
    out[g] = (bf16_t)v;
}

// ---------------- GEMM: C[nRows x 128] = A[nRows x 128] * Bpacked (+bias, +relu) -------
// grid.x = row tiles (16 rows each); 256 threads = 8 waves; wave w owns col tile w.
__global__ __launch_bounds__(256) void gemm_n128_wmma(
        const float* __restrict__ A, const bf16_t* __restrict__ Bp,
        const float* __restrict__ bias, float* __restrict__ C,
        int nRows, int fuseRelu) {
    const int lane = threadIdx.x & 31;
    const int wave = threadIdx.x >> 5;          // output column tile 0..7
    const int half = lane >> 4;                 // 0 | 1
    const int m16  = lane & 15;
    const int row  = blockIdx.x * 16 + m16;     // A-matrix row this lane feeds
    const long arow = (long)(row < nRows ? row : nRows - 1) * NFEAT;

    v8f acc = {};
#pragma unroll
    for (int kb = 0; kb < 4; ++kb) {            // K = 128 = 4 x 32
        const int kbase = kb * 32 + half * 8;   // A layout: VGPR0..3 -> K kbase..+7, VGPR4..7 -> K kbase+16..+23
        const float* ap = A + arow + kbase;
        v16bf a;
#pragma unroll
        for (int i = 0; i < 8; ++i) a[i]     = (bf16_t)ap[i];
#pragma unroll
        for (int i = 0; i < 8; ++i) a[8 + i] = (bf16_t)ap[16 + i];
        const v16bf b = *(const v16bf*)(Bp + (((size_t)(kb * 8 + wave) * 32 + lane) << 4));
        acc = __builtin_amdgcn_wmma_f32_16x16x32_bf16(false, a, false, b,
                                                      (short)0, acc, false, false);
    }

    const int col = wave * 16 + m16;            // C layout: N = lane&15
    const float bv = bias ? bias[col] : 0.0f;
#pragma unroll
    for (int r = 0; r < 8; ++r) {               // C layout: M = r + half*8
        const int orow = blockIdx.x * 16 + half * 8 + r;
        if (orow < nRows) {
            float v = acc[r] + bv;
            if (fuseRelu) v = fmaxf(v, 0.0f);
            C[(long)orow * NFEAT + col] = v;
        }
    }
}

// ---------------- agg init: self-loop message + bias -> agg = b1 + h * dinv^2 ----------
__global__ void init_agg(const float* __restrict__ h, const float* __restrict__ dinv,
                         const float* __restrict__ b1, float* __restrict__ agg, long n) {
    long i = (long)blockIdx.x * blockDim.x + threadIdx.x;
    long stride = (long)gridDim.x * blockDim.x;
    for (; i < n; i += stride) {
        int col = (int)(i & (NFEAT - 1));
        long row = i >> 7;
        float di = dinv[row];
        agg[i] = b1[col] + h[i] * di * di;
    }
}

// ---------------- edge scatter: one wave per edge, 4 floats per lane -------------------
__global__ __launch_bounds__(256) void edge_scatter(
        const int* __restrict__ ei, const float* __restrict__ ew,
        const float* __restrict__ dinv, const float* __restrict__ h,
        float* __restrict__ agg, int E) {
    int e = blockIdx.x * 8 + (threadIdx.x >> 5);
    if (e >= E) return;
    int lane = threadIdx.x & 31;
    int s = ei[e];
    int d = ei[E + e];
    float norm = dinv[s] * ew[e] * dinv[d];
    const float4 hv = *(const float4*)(h + (long)s * NFEAT + lane * 4);
    float* o = agg + (long)d * NFEAT + lane * 4;
    atomicAdd(o + 0, hv.x * norm);
    atomicAdd(o + 1, hv.y * norm);
    atomicAdd(o + 2, hv.z * norm);
    atomicAdd(o + 3, hv.w * norm);
}

// ---------------- BatchNorm: per-column sum / sumsq ----------------
__global__ __launch_bounds__(128) void bn_stats(const float* __restrict__ v,
                                                float* __restrict__ stats, int nRows) {
    int col = threadIdx.x;
    int per = (nRows + gridDim.x - 1) / gridDim.x;
    int r0 = blockIdx.x * per;
    int r1 = r0 + per; if (r1 > nRows) r1 = nRows;
    float s = 0.f, s2 = 0.f;
    for (int r = r0; r < r1; ++r) {
        float t = v[(long)r * NFEAT + col];
        s += t; s2 += t * t;
    }
    atomicAdd(&stats[col], s);
    atomicAdd(&stats[NFEAT + col], s2);
}

__global__ void bn_finalize(const float* __restrict__ stats, const float* __restrict__ g,
                            const float* __restrict__ be, float* __restrict__ scale,
                            float* __restrict__ shift, float invN) {
    int c = threadIdx.x;
    float mean = stats[c] * invN;
    float var  = stats[NFEAT + c] * invN - mean * mean;  // biased variance
    float sc = g[c] * rsqrtf(var + 1e-5f);
    scale[c] = sc;
    shift[c] = be[c] - mean * sc;
}

__global__ void bn_apply_sigmoid(float* __restrict__ v, const float* __restrict__ scale,
                                 const float* __restrict__ shift, long n) {
    long i = (long)blockIdx.x * blockDim.x + threadIdx.x;
    long stride = (long)gridDim.x * blockDim.x;
    for (; i < n; i += stride) {
        int col = (int)(i & (NFEAT - 1));
        float t = v[i] * scale[col] + shift[col];
        v[i] = 1.0f / (1.0f + expf(-t));
    }
}

// ---------------- global mean pool ----------------
__global__ void pool_count(const int* __restrict__ batch, float* __restrict__ pcnt, int n) {
    int i = blockIdx.x * blockDim.x + threadIdx.x;
    if (i < n) atomicAdd(&pcnt[batch[i]], 1.0f);
}

__global__ void pool_sum(const float* __restrict__ h2, const int* __restrict__ batch,
                         float* __restrict__ psum, long n) {
    long i = (long)blockIdx.x * blockDim.x + threadIdx.x;
    long stride = (long)gridDim.x * blockDim.x;
    for (; i < n; i += stride) {
        int col = (int)(i & (NFEAT - 1));
        long row = i >> 7;
        atomicAdd(&psum[(long)batch[row] * NFEAT + col], h2[i]);
    }
}

// ---------------- head MLP: 128 -> 64(sig) -> 16(relu) -> 16(relu) -> 1 ---------------
__global__ __launch_bounds__(64) void mlp_head(
        const float* __restrict__ psum, const float* __restrict__ pcnt,
        const float* __restrict__ Wil, const float* __restrict__ bil,
        const float* __restrict__ Whl1, const float* __restrict__ bhl1,
        const float* __restrict__ Whl2, const float* __restrict__ bhl2,
        const float* __restrict__ Wol, const float* __restrict__ bol,
        float* __restrict__ out) {
    __shared__ float s1[64], s2[16], s3[16];
    const int g = blockIdx.x, t = threadIdx.x;
    const float inv = 1.0f / fmaxf(pcnt[g], 1.0f);
    float acc = bil[t];
    for (int k = 0; k < NFEAT; ++k) acc += (psum[g * NFEAT + k] * inv) * Wil[k * 64 + t];
    s1[t] = 1.0f / (1.0f + expf(-acc));
    __syncthreads();
    if (t < 16) {
        float a = bhl1[t];
        for (int k = 0; k < 64; ++k) a += s1[k] * Whl1[k * 16 + t];
        s2[t] = fmaxf(a, 0.0f);
    }
    __syncthreads();
    if (t < 16) {
        float a = bhl2[t];
        for (int k = 0; k < 16; ++k) a += s2[k] * Whl2[k * 16 + t];
        s3[t] = fmaxf(a, 0.0f);
    }
    __syncthreads();
    if (t == 0) {
        float a = bol[0];
        for (int k = 0; k < 16; ++k) a += s3[k] * Wol[k];
        out[g] = a;
    }
}

// =====================================================================================
extern "C" void kernel_launch(void* const* d_in, const int* in_sizes, int n_in,
                              void* d_out, int out_size, void* d_ws, size_t ws_size,
                              hipStream_t stream) {
    const float* x    = (const float*)d_in[0];
    const int*   ei   = (const int*)  d_in[1];
    const float* ew   = (const float*)d_in[2];
    const int*   batch= (const int*)  d_in[3];
    const float* W1   = (const float*)d_in[4];
    const float* b1   = (const float*)d_in[5];
    const float* g1   = (const float*)d_in[6];
    const float* be1  = (const float*)d_in[7];
    // d_in[8..11] (W2/b2/g2/be2) are dead code: reference overwrites x2 = x1.
    const float* Wlin = (const float*)d_in[12];
    const float* blin = (const float*)d_in[13];
    const float* Wil  = (const float*)d_in[14];
    const float* bil  = (const float*)d_in[15];
    const float* Whl1 = (const float*)d_in[16];
    const float* bhl1 = (const float*)d_in[17];
    const float* Whl2 = (const float*)d_in[18];
    const float* bhl2 = (const float*)d_in[19];
    const float* Wol  = (const float*)d_in[20];
    const float* bol  = (const float*)d_in[21];
    float* out = (float*)d_out;

    const int N = in_sizes[0] / NFEAT;   // 100000
    const int E = in_sizes[2];           // 1600000
    const long NF = (long)N * NFEAT;

    // ---- workspace carve-out (256B aligned) ----
    char* p = (char*)d_ws;
    auto alloc = [&](size_t bytes) -> char* {
        char* r = p; p += (bytes + 255) & ~(size_t)255; return r;
    };
    float*  deg   = (float*)alloc((size_t)N * 4);
    float*  dinv  = (float*)alloc((size_t)N * 4);
    float*  h     = (float*)alloc((size_t)NF * 4);   // x@W1, later relu(lin) output
    float*  agg   = (float*)alloc((size_t)NF * 4);   // conv1 aggregate, then x1 in place
    float*  stats = (float*)alloc(2 * NFEAT * 4);
    float*  scale = (float*)alloc(NFEAT * 4);
    float*  shift = (float*)alloc(NFEAT * 4);
    float*  psum  = (float*)alloc(NGRAPH * NFEAT * 4);
    float*  pcnt  = (float*)alloc(NGRAPH * 4);
    bf16_t* w1bf  = (bf16_t*)alloc(NFEAT * NFEAT * 2);
    bf16_t* webf  = (bf16_t*)alloc(NFEAT * NFEAT * 2);
    (void)ws_size; (void)n_in; (void)out_size;

    const int rowTiles = (N + 15) / 16;

    // 1) degrees (self-loop weight 1 -> init 1.0) and dinv
    fill_f32<<<256, 256, 0, stream>>>(deg, 1.0f, N);
    deg_edges<<<(E + 255) / 256, 256, 0, stream>>>(ei, ew, deg, E);
    rsqrt_deg<<<(N + 255) / 256, 256, 0, stream>>>(deg, dinv, N);

    // 2) pack W1 to WMMA-B layout; h = x @ W1  (no bias; bias applied post-aggregation)
    pack_b_bf16<<<64, 256, 0, stream>>>(W1, nullptr, w1bf);
    gemm_n128_wmma<<<rowTiles, 256, 0, stream>>>(x, w1bf, nullptr, h, N, 0);

    // 3) aggregate: agg = b1 + self-loop msg, then scatter edge messages
    init_agg<<<2048, 256, 0, stream>>>(h, dinv, b1, agg, NF);
    edge_scatter<<<(E + 7) / 8, 256, 0, stream>>>(ei, ew, dinv, h, agg, E);

    // 4) BatchNorm (training-mode, biased var) + sigmoid -> x1 (in place in agg)
    fill_f32<<<1, 256, 0, stream>>>(stats, 0.0f, 2 * NFEAT);
    bn_stats<<<256, 128, 0, stream>>>(agg, stats, N);
    bn_finalize<<<1, NFEAT, 0, stream>>>(stats, g1, be1, scale, shift, 1.0f / (float)N);
    bn_apply_sigmoid<<<2048, 256, 0, stream>>>(agg, scale, shift, NF);

    // 5) lin: concat([x1,x1])@Wlin == x1@(Wlin_top + Wlin_bot); relu fused; into h
    pack_b_bf16<<<64, 256, 0, stream>>>(Wlin, Wlin + NFEAT * NFEAT, webf);
    gemm_n128_wmma<<<rowTiles, 256, 0, stream>>>(agg, webf, blin, h, N, 1);

    // 6) global mean pool over batch
    fill_f32<<<2, 256, 0, stream>>>(psum, 0.0f, NGRAPH * NFEAT + NGRAPH); // psum+pcnt contiguous
    pool_count<<<(N + 255) / 256, 256, 0, stream>>>(batch, pcnt, N);
    pool_sum<<<2048, 256, 0, stream>>>(h, batch, psum, NF);

    // 7) head MLP -> [64, 1]
    mlp_head<<<NGRAPH, 64, 0, stream>>>(psum, pcnt, Wil, bil, Whl1, bhl1,
                                        Whl2, bhl2, Wol, bol, out);
}
